// PNNStack_46445776339565
// MI455X (gfx1250) — matrix operand order repaired and checked by reading, not compiled, not run on hardware
//
#include <hip/hip_runtime.h>
#include <hip/hip_bf16.h>
#include <hip/hip_fp16.h>

typedef __attribute__((ext_vector_type(16))) _Float16 v16h;
typedef __attribute__((ext_vector_type(8)))  _Float16 v8h;
typedef __attribute__((ext_vector_type(8)))  float    v8f;
typedef __attribute__((ext_vector_type(4)))  unsigned u32x4;
typedef __attribute__((ext_vector_type(8)))  unsigned u32x8;

#define HIDW   50
#define TOW    5
#define FOUT   10
#define NGRP   64
#define EPS_BN 1e-5f
#define EPS_SD 1e-5f

static inline int cdiv(long long a, long long b) { return (int)((a + b - 1) / b); }

// ---- monotonic float<->int key for integer atomic min/max over floats ----
__device__ __forceinline__ int fkey(float f) {
    int i = __float_as_int(f);
    return (i >= 0) ? i : (i ^ 0x7FFFFFFF);
}
__device__ __forceinline__ float funkey(int m) {
    int i = (m >= 0) ? m : (m ^ 0x7FFFFFFF);
    return __int_as_float(i);
}

// ------------------------------ utility kernels ------------------------------
__global__ void k_fill_u32(unsigned int* p, unsigned int v, long long n) {
    long long i = (long long)blockIdx.x * blockDim.x + threadIdx.x;
    if (i < n) p[i] = v;
}

__global__ void k_degree(const int* __restrict__ ei, float* __restrict__ deg, int E) {
    int e = blockIdx.x * blockDim.x + threadIdx.x;
    if (e < E) atomicAdd(&deg[ei[E + e]], 1.0f);
}

__global__ void k_logsum(const float* __restrict__ deg, float* __restrict__ scal, int N) {
    int i = blockIdx.x * blockDim.x + threadIdx.x;
    if (i < N) atomicAdd(&scal[0], logf(deg[i] + 1.0f));
}

__global__ void k_amp(const float* __restrict__ deg, const float* __restrict__ scal,
                      float* __restrict__ amp, int N) {
    int i = blockIdx.x * blockDim.x + threadIdx.x;
    if (i < N) {
        float avg = scal[0] / (float)N;
        amp[i] = logf(fmaxf(deg[i], 1.0f) + 1.0f) / avg;
    }
}

__global__ void k_f32_to_f16(const float* __restrict__ src, _Float16* __restrict__ dst, long long n) {
    long long i = (long long)blockIdx.x * blockDim.x + threadIdx.x;
    if (i < n) dst[i] = (_Float16)src[i];
}

// pack pre_W [NC, K] f32 -> [NCPAD, KPAD] f16 (zero padded), row = output col, contiguous K
__global__ void k_pack_w(const float* __restrict__ preW, _Float16* __restrict__ wp,
                         int NC, int NCPAD, int K, int KPAD) {
    int idx = blockIdx.x * blockDim.x + threadIdx.x;
    if (idx < NCPAD * KPAD) {
        int col = idx / KPAD, k = idx - col * KPAD;
        float v = (col < NC && k < K) ? preW[(long long)col * K + k] : 0.0f;
        wp[idx] = (_Float16)v;
    }
}

// ------------------- edge message GEMM (WMMA) + fused scatter -------------------
// Weights DMA'd global->LDS by the Tensor Data Mover (one contiguous tile,
// data_size = 8B, 1 row). A tile: 256 edges x KPAD f16, row = cat(x[dst],x[src],0).
// Output D[m][n] scattered with 4 aggregate atomics into L2-resident node buffers.
template<int F, int NC, int NCPAD, int KPAD>
__global__ __launch_bounds__(256) void k_edge_gemm_scatter(
    const _Float16* __restrict__ xh, const _Float16* __restrict__ wp,
    const float* __restrict__ bias, const int* __restrict__ ei, int E,
    float* __restrict__ osum, float* __restrict__ osum2,
    int* __restrict__ omin, int* __restrict__ omax)
{
    extern __shared__ __align__(16) char smem[];
    _Float16* sW   = (_Float16*)smem;              // NCPAD * KPAD f16
    _Float16* sA   = sW + NCPAD * KPAD;            // 256 * KPAD f16
    int*      sDst = (int*)(sA + 256 * KPAD);      // 256 dst indices

    const int tid = threadIdx.x;
    const int blockBase = blockIdx.x * 256;

    // ---- TDM: DMA packed weight tile into LDS (wave 0 issues; TENSORcnt tracked) ----
    if (tid < 32) {
        const unsigned tot8 = (unsigned)(NCPAD * KPAD / 4);   // bytes/8: 8B-unit elements
        unsigned long long ga = (unsigned long long)(const void*)wp;
        unsigned ldsOff = (unsigned)(unsigned long long)(const void*)sW; // addr[31:0] == LDS offset
        // D# group0: count=1 | lds_addr | global_addr[56:0] | type=2 (bits 127:126)
        u32x4 g0 = { 1u,
                     ldsOff,
                     (unsigned)ga,
                     ((unsigned)(ga >> 32) & 0x01FFFFFFu) | 0x80000000u };
        // D# group1: data_size=3 (8B); tensor_dim0 = tile_dim0 = tot8; dim1 = 1; stride0 = tot8
        u32x8 g1 = { 0x00030000u,
                     (tot8 & 0xFFFFu) << 16,
                     ((tot8 >> 16) & 0xFFFFu) | (1u << 16),
                     (tot8 & 0xFFFFu) << 16,
                     1u,
                     tot8,
                     0u,
                     0u };
        u32x4 gz = { 0u, 0u, 0u, 0u };               // groups 2/3: zero (dims unused)
        asm volatile("tensor_load_to_lds %0, %1, %2, %3"
                     :: "s"(g0), "s"(g1), "s"(gz), "s"(gz)
                     : "memory");
        __builtin_amdgcn_s_wait_tensorcnt(0);
    }

    // one edge per thread: build cat(x[dst], x[src]) row, zero K pad; cache dst index
    {
        int e = blockBase + tid;
        _Float16* row = sA + tid * KPAD;
        if (e < E) {
            int s = ei[e], d = ei[E + e];
            sDst[tid] = d;
            const _Float16* xd = xh + (long long)d * F;
            const _Float16* xs = xh + (long long)s * F;
            for (int k = 0; k < F; ++k) { row[k] = xd[k]; row[F + k] = xs[k]; }
            for (int k = 2 * F; k < KPAD; ++k) row[k] = (_Float16)0.0f;
        } else {
            sDst[tid] = 0;
            for (int k = 0; k < KPAD; ++k) row[k] = (_Float16)0.0f;
        }
    }
    __syncthreads();

    const int lane = tid & 31;                // wave32
    const int wave = tid >> 5;
    const int hiHalf = (lane & 16) ? 1 : 0;
    const int l15 = lane & 15;
    const int NT = NCPAD / 16;

    for (int tile = wave; tile < 16 * NT; tile += 8) {
        const int mt = tile / NT, nt = tile % NT;
        v8f acc = {0.f, 0.f, 0.f, 0.f, 0.f, 0.f, 0.f, 0.f};
#pragma unroll
        for (int ks = 0; ks < KPAD / 32; ++ks) {
            const int K0 = ks * 32;
            // A fragment (16x32 f16): lanes0-15 K{0..7,16..23}; lanes16-31 K{8..15,24..31}
            const _Float16* pa = sA + (mt * 16 + l15) * KPAD + K0 + (hiHalf ? 8 : 0);
            v8h alo = *(const v8h*)pa;
            v8h ahi = *(const v8h*)(pa + 16);
            v16h a;
#pragma unroll
            for (int i = 0; i < 8; ++i) { a[i] = alo[i]; a[i + 8] = ahi[i]; }
            // B fragment (32x16 f16): lanes0-15 K{0..7,16..23}; lanes16-31 K{8..15,24..31}, col = lane&15
            const _Float16* pb = sW + (nt * 16 + l15) * KPAD + K0 + (hiHalf ? 8 : 0);
            v8h blo = *(const v8h*)pb;
            v8h bhi = *(const v8h*)(pb + 16);
            v16h b;
#pragma unroll
            for (int i = 0; i < 8; ++i) { b[i] = blo[i]; b[i + 8] = bhi[i]; }
            acc = __builtin_amdgcn_wmma_f32_16x16x32_f16(
                false, a, false, b, (short)0, acc, false, false);
        }
        const int col = nt * 16 + l15;
        if (col < NC) {
            const float bv = bias[col];
#pragma unroll
            for (int j = 0; j < 8; ++j) {
                int m = (hiHalf ? 8 : 0) + j;
                int e = blockBase + mt * 16 + m;
                if (e < E) {
                    int d = sDst[mt * 16 + m];
                    float v = acc[j] + bv;
                    long long o = (long long)d * NC + col;
                    atomicAdd(&osum[o], v);
                    atomicAdd(&osum2[o], v * v);
                    int key = fkey(v);
                    atomicMin(&omin[o], key);
                    atomicMax(&omax[o], key);
                }
            }
        }
    }
}

// ------------------- per-node: scalers + post-MLP + lin -------------------
template<int F>
__global__ __launch_bounds__(128) void k_node_post(
    const float* __restrict__ xin, const float* __restrict__ deg,
    const float* __restrict__ amp,
    const float* __restrict__ osum, const float* __restrict__ osum2,
    const int* __restrict__ omin, const int* __restrict__ omax,
    const float* __restrict__ postW, const float* __restrict__ postB,
    const float* __restrict__ linW, const float* __restrict__ linB,
    float* __restrict__ out)
{
    const int n = blockIdx.x;
    const int NC = TOW * F;
    const int TF13 = 13 * F;
    __shared__ float sfeat[TOW * 13 * HIDW];
    __shared__ float stow[TOW * FOUT];

    const float dg = deg[n];
    const float degc = fmaxf(dg, 1.0f);
    const float a = amp[n], ia = 1.0f / a;
    const bool has = dg > 0.0f;

    for (int tf = threadIdx.x; tf < NC; tf += blockDim.x) {
        int t = tf / F, f = tf - t * F;
        long long o = (long long)n * NC + tf;
        float s = osum[o], s2 = osum2[o];
        float mean = s / degc;
        float var = s2 / degc - mean * mean;
        float sd = sqrtf(fmaxf(var, 0.0f) + EPS_SD);
        float mn = has ? funkey(omin[o]) : 0.0f;
        float mx = has ? funkey(omax[o]) : 0.0f;
        float* fb = sfeat + t * TF13;
        fb[0 * F + f] = xin[(long long)n * F + f];
        fb[1 * F + f] = mean; fb[2 * F + f] = mn;      fb[3 * F + f] = mx;      fb[4 * F + f] = sd;
        fb[5 * F + f] = mean * a;  fb[6 * F + f] = mn * a;  fb[7 * F + f] = mx * a;  fb[8 * F + f] = sd * a;
        fb[9 * F + f] = mean * ia; fb[10 * F + f] = mn * ia; fb[11 * F + f] = mx * ia; fb[12 * F + f] = sd * ia;
    }
    __syncthreads();

    for (int o = threadIdx.x; o < TOW * FOUT; o += blockDim.x) {
        int t = o / FOUT;
        const float* w = postW + (long long)o * TF13;
        const float* fb = sfeat + t * TF13;
        float accv = postB[o];
        for (int j = 0; j < TF13; ++j) accv += fb[j] * w[j];
        stow[o] = accv;
    }
    __syncthreads();

    for (int c = threadIdx.x; c < HIDW; c += blockDim.x) {
        const float* w = linW + c * HIDW;
        float accv = linB[c];
#pragma unroll
        for (int k = 0; k < HIDW; ++k) accv += stow[k] * w[k];
        out[(long long)n * HIDW + c] = accv;
    }
}

// ------------------------------- batch norm -------------------------------
__global__ __launch_bounds__(256) void k_bn_stats(const float* __restrict__ h,
                                                  float* __restrict__ mu,
                                                  float* __restrict__ var, int N) {
    const int c = blockIdx.x;
    __shared__ float s1[256], s2[256];
    float a = 0.f, b = 0.f;
    for (int n = threadIdx.x; n < N; n += 256) {
        float v = h[(long long)n * HIDW + c];
        a += v; b += v * v;
    }
    s1[threadIdx.x] = a; s2[threadIdx.x] = b;
    __syncthreads();
    for (int st = 128; st > 0; st >>= 1) {
        if (threadIdx.x < st) { s1[threadIdx.x] += s1[threadIdx.x + st]; s2[threadIdx.x] += s2[threadIdx.x + st]; }
        __syncthreads();
    }
    if (threadIdx.x == 0) {
        float m = s1[0] / (float)N;
        mu[c] = m;
        var[c] = s2[0] / (float)N - m * m;
    }
}

__global__ void k_bn_apply(float* __restrict__ h, const float* __restrict__ mu,
                           const float* __restrict__ var, const float* __restrict__ gamma,
                           const float* __restrict__ beta, _Float16* __restrict__ xh_out,
                           long long total) {
    long long i = (long long)blockIdx.x * blockDim.x + threadIdx.x;
    if (i < total) {
        int c = (int)(i % HIDW);
        float v = h[i];
        v = gamma[c] * (v - mu[c]) * rsqrtf(var[c] + EPS_BN) + beta[c];
        v = fmaxf(v, 0.0f);
        h[i] = v;
        if (xh_out) xh_out[i] = (_Float16)v;
    }
}

// -------------------------------- pooling + head --------------------------------
__global__ void k_pool(const float* __restrict__ h, const int* __restrict__ batch,
                       float* __restrict__ pool, float* __restrict__ cnt, int N) {
    long long i = (long long)blockIdx.x * blockDim.x + threadIdx.x;
    if (i < (long long)N * HIDW) {
        int n = (int)(i / HIDW), c = (int)(i % HIDW);
        int g = batch[n];
        atomicAdd(&pool[g * HIDW + c], h[i]);
        if (c == 0) atomicAdd(&cnt[g], 1.0f);
    }
}

__global__ __launch_bounds__(256) void k_head(
    const float* __restrict__ pool, const float* __restrict__ cnt,
    const float* __restrict__ W1, const float* __restrict__ b1,
    const float* __restrict__ W2, const float* __restrict__ b2,
    const float* __restrict__ W3, const float* __restrict__ b3,
    float* __restrict__ out)
{
    __shared__ float sp[NGRP * HIDW], sh1[NGRP * HIDW], sh2[NGRP * 25];
    for (int i = threadIdx.x; i < NGRP * HIDW; i += 256) {
        int g = i / HIDW;
        sp[i] = pool[i] / fmaxf(cnt[g], 1.0f);
    }
    __syncthreads();
    for (int i = threadIdx.x; i < NGRP * HIDW; i += 256) {
        int g = i / HIDW, j = i % HIDW;
        float acc = b1[j];
        for (int k = 0; k < HIDW; ++k) acc += sp[g * HIDW + k] * W1[j * HIDW + k];
        sh1[i] = fmaxf(acc, 0.0f);
    }
    __syncthreads();
    for (int i = threadIdx.x; i < NGRP * 25; i += 256) {
        int g = i / 25, j = i % 25;
        float acc = b2[j];
        for (int k = 0; k < HIDW; ++k) acc += sh1[g * HIDW + k] * W2[j * HIDW + k];
        sh2[i] = fmaxf(acc, 0.0f);
    }
    __syncthreads();
    for (int g = threadIdx.x; g < NGRP; g += 256) {
        float acc = b3[0];
        for (int k = 0; k < 25; ++k) acc += sh2[g * 25 + k] * W3[k];
        out[g] = acc;
    }
}

// ------------------------------------ launcher ------------------------------------
extern "C" void kernel_launch(void* const* d_in, const int* in_sizes, int n_in,
                              void* d_out, int out_size, void* d_ws, size_t ws_size,
                              hipStream_t stream) {
    const float* x      = (const float*)d_in[0];
    const int*   ei     = (const int*)d_in[1];
    const int*   batch  = (const int*)d_in[2];
    const float* preW0  = (const float*)d_in[3];
    const float* preB0  = (const float*)d_in[4];
    const float* postW0 = (const float*)d_in[5];
    const float* postB0 = (const float*)d_in[6];
    const float* linW0  = (const float*)d_in[7];
    const float* linB0  = (const float*)d_in[8];
    const float* gamma0 = (const float*)d_in[9];
    const float* beta0  = (const float*)d_in[10];
    const float* preW1  = (const float*)d_in[11];
    const float* preB1  = (const float*)d_in[12];
    const float* postW1 = (const float*)d_in[13];
    const float* postB1 = (const float*)d_in[14];
    const float* linW1  = (const float*)d_in[15];
    const float* linB1  = (const float*)d_in[16];
    const float* gamma1 = (const float*)d_in[17];
    const float* beta1  = (const float*)d_in[18];
    const float* mW1    = (const float*)d_in[19];
    const float* mB1    = (const float*)d_in[20];
    const float* mW2    = (const float*)d_in[21];
    const float* mB2    = (const float*)d_in[22];
    const float* mW3    = (const float*)d_in[23];
    const float* mB3    = (const float*)d_in[24];

    const int N = in_sizes[0] / 16;   // 20000
    const int E = in_sizes[1] / 2;    // 320000

    // ---- workspace carve ----
    char* wsp = (char*)d_ws;
    auto carve = [&](size_t bytes) -> void* {
        void* r = (void*)wsp;
        wsp += (bytes + 255) & ~(size_t)255;
        return r;
    };
    float*    deg   = (float*)carve((size_t)N * 4);
    float*    amp   = (float*)carve((size_t)N * 4);
    float*    scal  = (float*)carve(256);
    float*    mu    = (float*)carve(256);
    float*    var   = (float*)carve(256);
    _Float16* xh    = (_Float16*)carve((size_t)N * 64 * 2);
    _Float16* wpack = (_Float16*)carve((size_t)256 * 128 * 2);
    float*    asum  = (float*)carve((size_t)N * 256 * 4);
    float*    asum2 = (float*)carve((size_t)N * 256 * 4);
    int*      amin  = (int*)carve((size_t)N * 256 * 4);
    int*      amax  = (int*)carve((size_t)N * 256 * 4);
    float*    hA    = (float*)carve((size_t)N * HIDW * 4);
    float*    hB    = (float*)carve((size_t)N * HIDW * 4);
    float*    pool  = (float*)carve((size_t)NGRP * HIDW * 4);
    float*    cnt   = (float*)carve((size_t)NGRP * 4);

    float* outp = (float*)d_out;

    // ---- degree + PNA scalers ----
    k_fill_u32<<<cdiv(N, 256), 256, 0, stream>>>((unsigned int*)deg, 0u, N);
    k_fill_u32<<<1, 64, 0, stream>>>((unsigned int*)scal, 0u, 4);
    k_degree<<<cdiv(E, 256), 256, 0, stream>>>(ei, deg, E);
    k_logsum<<<cdiv(N, 256), 256, 0, stream>>>(deg, scal, N);
    k_amp<<<cdiv(N, 256), 256, 0, stream>>>(deg, scal, amp, N);

    // =================== conv layer 0 (F=16, NC=80, K=32) ===================
    {
        const int F = 16, NC = 80, NCPAD = 80, K = 32, KPAD = 32;
        k_f32_to_f16<<<cdiv((long long)N * F, 256), 256, 0, stream>>>(x, xh, (long long)N * F);
        k_pack_w<<<cdiv(NCPAD * KPAD, 256), 256, 0, stream>>>(preW0, wpack, NC, NCPAD, K, KPAD);
        long long agg = (long long)N * NC;
        k_fill_u32<<<cdiv(agg, 256), 256, 0, stream>>>((unsigned int*)asum, 0u, agg);
        k_fill_u32<<<cdiv(agg, 256), 256, 0, stream>>>((unsigned int*)asum2, 0u, agg);
        k_fill_u32<<<cdiv(agg, 256), 256, 0, stream>>>((unsigned int*)amin, 0x7FFFFFFFu, agg);
        k_fill_u32<<<cdiv(agg, 256), 256, 0, stream>>>((unsigned int*)amax, 0x80000000u, agg);
        size_t lds = (size_t)(NCPAD * KPAD + 256 * KPAD) * sizeof(_Float16) + 256 * sizeof(int);
        k_edge_gemm_scatter<F, NC, NCPAD, KPAD><<<cdiv(E, 256), 256, lds, stream>>>(
            xh, wpack, preB0, ei, E, asum, asum2, amin, amax);
        k_node_post<F><<<N, 128, 0, stream>>>(x, deg, amp, asum, asum2, amin, amax,
                                              postW0, postB0, linW0, linB0, hA);
        k_bn_stats<<<HIDW, 256, 0, stream>>>(hA, mu, var, N);
        k_bn_apply<<<cdiv((long long)N * HIDW, 256), 256, 0, stream>>>(
            hA, mu, var, gamma0, beta0, xh, (long long)N * HIDW);  // also f16 for next layer
    }

    // =================== conv layer 1 (F=50, NC=250, K=100->128) ===================
    {
        const int F = 50, NC = 250, NCPAD = 256, K = 100, KPAD = 128;
        k_pack_w<<<cdiv(NCPAD * KPAD, 256), 256, 0, stream>>>(preW1, wpack, NC, NCPAD, K, KPAD);
        long long agg = (long long)N * NC;
        k_fill_u32<<<cdiv(agg, 256), 256, 0, stream>>>((unsigned int*)asum, 0u, agg);
        k_fill_u32<<<cdiv(agg, 256), 256, 0, stream>>>((unsigned int*)asum2, 0u, agg);
        k_fill_u32<<<cdiv(agg, 256), 256, 0, stream>>>((unsigned int*)amin, 0x7FFFFFFFu, agg);
        k_fill_u32<<<cdiv(agg, 256), 256, 0, stream>>>((unsigned int*)amax, 0x80000000u, agg);
        size_t lds = (size_t)(NCPAD * KPAD + 256 * KPAD) * sizeof(_Float16) + 256 * sizeof(int); // ~129 KB
        k_edge_gemm_scatter<F, NC, NCPAD, KPAD><<<cdiv(E, 256), 256, lds, stream>>>(
            xh, wpack, preB1, ei, E, asum, asum2, amin, amax);
        k_node_post<F><<<N, 128, 0, stream>>>(hA, deg, amp, asum, asum2, amin, amax,
                                              postW1, postB1, linW1, linB1, hB);
        k_bn_stats<<<HIDW, 256, 0, stream>>>(hB, mu, var, N);
        k_bn_apply<<<cdiv((long long)N * HIDW, 256), 256, 0, stream>>>(
            hB, mu, var, gamma1, beta1, (_Float16*)nullptr, (long long)N * HIDW);
    }

    // =================== pooling + MLP head ===================
    k_fill_u32<<<cdiv(NGRP * HIDW, 256), 256, 0, stream>>>((unsigned int*)pool, 0u, NGRP * HIDW);
    k_fill_u32<<<1, 64, 0, stream>>>((unsigned int*)cnt, 0u, NGRP);
    k_pool<<<cdiv((long long)N * HIDW, 256), 256, 0, stream>>>(hB, batch, pool, cnt, N);
    k_head<<<1, 256, 0, stream>>>(pool, cnt, mW1, mB1, mW2, mB2, mW3, mB3, outp);
}
